// OpenPanguMLAAttention_29566554865741
// MI455X (gfx1250) — matrix operand order repaired and checked by reading, not compiled, not run on hardware
//
#include <hip/hip_runtime.h>

// ---------------------------------------------------------------------------
// MLA attention prefill for MI455X (gfx1250, wave32, WMMA).
// Heavy math: v_wmma_f32_16x16x32_bf16 (f32 accumulate). GEMMs use a
// double-buffered LDS pipeline with global_load_async_to_lds_b128 staging
// (ASYNCcnt) for the contiguous A tiles. softmax/rmsnorm/RoPE stay in f32.
// ---------------------------------------------------------------------------

typedef __bf16 bf16_t;
typedef __bf16  v16bf __attribute__((ext_vector_type(16)));
typedef float   v8f   __attribute__((ext_vector_type(8)));

#define T_LEN   2048
#define HIDDEN  5120
#define NH      32
#define DQ      128
#define DR      64
#define DV      128
#define Q_LORA  1536
#define KV_LORA 512
#define DQR     192   // DQ + DR

union Frag { uint4 u[2]; v16bf v; };

__device__ __forceinline__ v8f wmma_bf16(v16bf a, v16bf b, v8f c) {
  return __builtin_amdgcn_wmma_f32_16x16x32_bf16(false, a, false, b, (short)0, c,
                                                 false, false);
}

// CDNA5 async copy: 16 bytes global -> LDS, tracked by ASYNCcnt.
__device__ __forceinline__ void async_ld16(unsigned lds_byte, const bf16_t* g) {
  asm volatile("global_load_async_to_lds_b128 %0, %1, off"
               :: "v"(lds_byte), "v"(g) : "memory");
}
__device__ __forceinline__ void wait_async0() {
  asm volatile("s_wait_asynccnt 0" ::: "memory");
}
__device__ __forceinline__ unsigned lds_addr(const void* p) {
  return (unsigned)(size_t)p;  // flat addr[31:0] == LDS byte offset (ISA 10.2)
}

// ---------------------------------------------------------------------------
// fp32 -> bf16 bulk convert (n multiple of 8)
// ---------------------------------------------------------------------------
__global__ __launch_bounds__(256)
void cvt_f32_bf16(const float* __restrict__ s, bf16_t* __restrict__ d, long n) {
  long i = ((long)blockIdx.x * 256 + threadIdx.x) * 8;
  if (i >= n) return;
  float4 a = *(const float4*)(s + i);
  float4 b = *(const float4*)(s + i + 4);
  union { bf16_t h[8]; uint4 u; } r;
  r.h[0] = (bf16_t)a.x; r.h[1] = (bf16_t)a.y; r.h[2] = (bf16_t)a.z; r.h[3] = (bf16_t)a.w;
  r.h[4] = (bf16_t)b.x; r.h[5] = (bf16_t)b.y; r.h[6] = (bf16_t)b.z; r.h[7] = (bf16_t)b.w;
  *(uint4*)(d + i) = r.u;
}

// ---------------------------------------------------------------------------
// bf16 GEMM: C(f32, MxN) = A(bf16, MxK row-major) * B(bf16, KxN row-major)
// BM=128, BN=64, BK=64, double-buffered LDS, async A staging.
// 256 threads = 8 waves; wave tile 32x32 (2x2 WMMA), 8 WMMAs per stage.
// Requires M%128==0, N%64==0, K%64==0 (true for all five GEMMs here).
// ---------------------------------------------------------------------------
__global__ __launch_bounds__(256)
void gemm_bf16_f32(const bf16_t* __restrict__ A, const bf16_t* __restrict__ B,
                   float* __restrict__ C, int M, int N, int K) {
  __shared__ __attribute__((aligned(16))) bf16_t ldsA[2][128 * 64];  // row stride 64
  __shared__ __attribute__((aligned(16))) bf16_t ldsBt[2][64 * 64];  // Bt[n][k]

  const int tid   = threadIdx.x;
  const int wid   = tid >> 5;
  const int lane  = tid & 31;
  const int laneL = lane & 15;
  const int hi    = lane >> 4;
  const int m0 = blockIdx.y * 128;
  const int n0 = blockIdx.x * 64;
  const int wm = (wid >> 1) * 32;
  const int wn = (wid & 1) * 32;

  // per-thread staging coordinates
  const int arow = tid >> 1;              // 0..127
  const int acol = (tid & 1) * 32;        // 0 / 32
  const int bk   = tid >> 2;              // 0..63
  const int bnc  = (tid & 3) * 16;        // 0/16/32/48

  auto stage = [&](int kt, int buf) {
    // A tile 128x64: 4 async b128 per thread, straight into LDS
    const bf16_t* ga = A + (long)(m0 + arow) * K + kt * 64 + acol;
    unsigned la = lds_addr(&ldsA[buf][arow * 64 + acol]);
#pragma unroll
    for (int c = 0; c < 4; ++c) async_ld16(la + 16 * c, ga + 8 * c);
    // B tile 64x64: coalesced row reads, transposed stores Bt[n][k]
    const bf16_t* gb = B + (long)(kt * 64 + bk) * N + n0 + bnc;
    union { uint4 u; bf16_t h[8]; } t0, t1;
    t0.u = *(const uint4*)gb;
    t1.u = *(const uint4*)(gb + 8);
#pragma unroll
    for (int e = 0; e < 8; ++e) {
      ldsBt[buf][(bnc + e) * 64 + bk]     = t0.h[e];
      ldsBt[buf][(bnc + 8 + e) * 64 + bk] = t1.h[e];
    }
  };

  v8f acc[2][2] = {};
  const int nt = K / 64;
  stage(0, 0);

  for (int kt = 0; kt < nt; ++kt) {
    const int cur = kt & 1;
    wait_async0();
    __syncthreads();
    if (kt + 1 < nt) stage(kt + 1, cur ^ 1);

#pragma unroll
    for (int s = 0; s < 2; ++s) {
      const int ks = s * 32;
      Frag fa[2], fb[2];
#pragma unroll
      for (int i = 0; i < 2; ++i) {
        int row = wm + i * 16 + laneL;
        fa[i].u[0] = *(const uint4*)&ldsA[cur][row * 64 + ks + hi * 8];
        fa[i].u[1] = *(const uint4*)&ldsA[cur][row * 64 + ks + 16 + hi * 8];
      }
#pragma unroll
      for (int j = 0; j < 2; ++j) {
        int col = wn + j * 16 + laneL;
        fb[j].u[0] = *(const uint4*)&ldsBt[cur][col * 64 + ks + hi * 8];
        fb[j].u[1] = *(const uint4*)&ldsBt[cur][col * 64 + ks + 16 + hi * 8];
      }
#pragma unroll
      for (int i = 0; i < 2; ++i)
#pragma unroll
        for (int j = 0; j < 2; ++j)
          acc[i][j] = wmma_bf16(fa[i].v, fb[j].v, acc[i][j]);
    }
  }

#pragma unroll
  for (int i = 0; i < 2; ++i)
#pragma unroll
    for (int j = 0; j < 2; ++j) {
      int colg = n0 + wn + j * 16 + laneL;
#pragma unroll
      for (int r = 0; r < 8; ++r) {
        int rowg = m0 + wm + i * 16 + hi * 8 + r;
        C[(long)rowg * N + colg] = acc[i][j][r];
      }
    }
}

// ---------------------------------------------------------------------------
// block reduce helper (256 threads, wave32)
// ---------------------------------------------------------------------------
__device__ __forceinline__ float block_sum(float v, float* red) {
#pragma unroll
  for (int off = 16; off > 0; off >>= 1) v += __shfl_xor(v, off, 32);
  int wid = threadIdx.x >> 5, ln = threadIdx.x & 31;
  if (ln == 0) red[wid] = v;
  __syncthreads();
  if (wid == 0) {
    float t = (ln < 8) ? red[ln] : 0.0f;
#pragma unroll
    for (int off = 4; off > 0; off >>= 1) t += __shfl_xor(t, off, 32);
    if (ln == 0) red[0] = t;
  }
  __syncthreads();
  return red[0];
}

// ---------------------------------------------------------------------------
// rmsnorm(x) * gamma -> bf16   (one block per row)
// ---------------------------------------------------------------------------
__global__ __launch_bounds__(256)
void rmsnorm_bf16_kernel(const float* __restrict__ x, const float* __restrict__ gamma,
                         bf16_t* __restrict__ out, int width) {
  __shared__ float red[8];
  int row = blockIdx.x;
  const float* xr = x + (long)row * width;
  float s = 0.f;
  for (int i = threadIdx.x; i < width; i += 256) { float v = xr[i]; s += v * v; }
  float tot = block_sum(s, red);
  float inv = rsqrtf(tot / (float)width + 1e-6f);
  bf16_t* o = out + (long)row * width;
  for (int i = threadIdx.x; i < width; i += 256) o[i] = (bf16_t)(xr[i] * inv * gamma[i]);
}

__device__ __forceinline__ void fill_rope(const int* pos, int t, float* cosl, float* sinl) {
  if (threadIdx.x < 64) {
    int j = threadIdx.x;
    float invf = __powf(10000.0f, -(float)(2 * (j & 31)) / 64.0f);
    float fr = (float)pos[t] * invf;
    cosl[j] = __cosf(fr);
    sinl[j] = __sinf(fr);
  }
}

// ---------------------------------------------------------------------------
// q finalize: qraw f32 (t, NH, 192) -> q_full bf16, RoPE on last 64 dims
// ---------------------------------------------------------------------------
__global__ __launch_bounds__(256)
void q_finalize_kernel(const float* __restrict__ qraw, const int* __restrict__ pos,
                       bf16_t* __restrict__ qfull) {
  __shared__ float cosl[64], sinl[64];
  int t = blockIdx.x;
  fill_rope(pos, t, cosl, sinl);
  __syncthreads();
  const float* row = qraw + (long)t * (NH * DQR);
  bf16_t* o = qfull + (long)t * (NH * DQR);
  for (int idx = threadIdx.x; idx < NH * DQ; idx += 256) {
    int h = idx >> 7, d = idx & 127;
    o[h * DQR + d] = (bf16_t)row[h * DQR + d];
  }
  for (int p = threadIdx.x; p < NH * (DR / 2); p += 256) {
    int h = p >> 5, i = p & 31, j0 = 2 * i, j1 = j0 + 1;
    float x0 = row[h * DQR + DQ + j0], x1 = row[h * DQR + DQ + j1];
    o[h * DQR + DQ + j0] = (bf16_t)(x0 * cosl[j0] - x1 * sinl[j0]);
    o[h * DQR + DQ + j1] = (bf16_t)(x1 * cosl[j1] + x0 * sinl[j1]);
  }
}

// ---------------------------------------------------------------------------
// kv finalize: kvraw f32 (t, 576) -> kv_c bf16 (rmsnorm) + k_pe RoPE broadcast
// into k_full[t][h][128..191] for all heads
// ---------------------------------------------------------------------------
__global__ __launch_bounds__(256)
void kv_finalize_kernel(const float* __restrict__ kvraw, const float* __restrict__ gamma,
                        const int* __restrict__ pos, bf16_t* __restrict__ kvc,
                        bf16_t* __restrict__ kfull) {
  __shared__ float red[8];
  __shared__ float cosl[64], sinl[64];
  int t = blockIdx.x;
  fill_rope(pos, t, cosl, sinl);
  const float* row = kvraw + (long)t * (KV_LORA + DR);
  float s = 0.f;
  for (int i = threadIdx.x; i < KV_LORA; i += 256) { float v = row[i]; s += v * v; }
  float tot = block_sum(s, red);
  float inv = rsqrtf(tot / (float)KV_LORA + 1e-6f);
  for (int i = threadIdx.x; i < KV_LORA; i += 256)
    kvc[(long)t * KV_LORA + i] = (bf16_t)(row[i] * inv * gamma[i]);
  __syncthreads();
  if (threadIdx.x < 32) {
    int i = threadIdx.x, j0 = 2 * i, j1 = j0 + 1;
    float x0 = row[KV_LORA + j0], x1 = row[KV_LORA + j1];
    bf16_t b0 = (bf16_t)(x0 * cosl[j0] - x1 * sinl[j0]);
    bf16_t b1 = (bf16_t)(x1 * cosl[j1] + x0 * sinl[j1]);
    for (int h = 0; h < NH; ++h) {
      kfull[((long)t * NH + h) * DQR + DQ + j0] = b0;
      kfull[((long)t * NH + h) * DQR + DQ + j1] = b1;
    }
  }
}

// ---------------------------------------------------------------------------
// kv up-projection split: kvup f32 (t, NH, 256) -> k_full nope slice + V
// ---------------------------------------------------------------------------
__global__ __launch_bounds__(256)
void kvup_finalize_kernel(const float* __restrict__ kvup, bf16_t* __restrict__ kfull,
                          bf16_t* __restrict__ v) {
  long i = (long)blockIdx.x * 256 + threadIdx.x;
  long n = (long)T_LEN * NH * (DQ + DV);
  if (i >= n) return;
  int d = (int)(i & 255);
  long th = i >> 8;  // t*NH + h
  float val = kvup[i];
  if (d < DQ) kfull[th * DQR + d] = (bf16_t)val;
  else        v[th * DV + (d - DQ)] = (bf16_t)val;
}

// ---------------------------------------------------------------------------
// Flash attention, one head x 64 queries per block. 4 waves x 16 queries.
// S = Q K^T via WMMA (6 k-steps of 32 over d=192), online softmax in f32,
// P -> LDS (bf16) -> WMMA P*V (2 k-steps of 32 over 64 keys).
// K tile staged with async-to-LDS.
// ---------------------------------------------------------------------------
__global__ __launch_bounds__(128)
void mla_attn_kernel(const bf16_t* __restrict__ Q,   // (T, NH, 192)
                     const bf16_t* __restrict__ Kf,  // (T, NH, 192)
                     const bf16_t* __restrict__ V,   // (T, NH*128)
                     bf16_t* __restrict__ O) {       // (T, NH*128)
  __shared__ __attribute__((aligned(16))) bf16_t sK[64 * DQR];     // 24 KB
  __shared__ __attribute__((aligned(16))) bf16_t sVt[DV * 64];     // 16 KB, Vt[d][k]
  __shared__ __attribute__((aligned(16))) bf16_t sP[4][16 * 64];   //  8 KB

  const int h  = blockIdx.y;
  const int q0 = blockIdx.x * 64;
  const int tid = threadIdx.x, wid = tid >> 5, lane = tid & 31;
  const int laneL = lane & 15, hiL = lane >> 4;
  const float scale = rsqrtf((float)DQR);

  v8f oacc[8] = {};
  float mstate[8], lstate[8];
#pragma unroll
  for (int r = 0; r < 8; ++r) { mstate[r] = -1e30f; lstate[r] = 0.f; }

  const int nkb = blockIdx.x + 1;  // causal: key blocks 0..q0/64
  for (int kb = 0; kb < nkb; ++kb) {
    const int k0 = kb * 64;
    __syncthreads();
    // stage K tile (64 x 192) via async-to-LDS (contiguous rows)
    for (int idx = tid * 16; idx < 64 * DQR; idx += 128 * 16) {
      int r = idx / DQR, c = idx % DQR;
      async_ld16(lds_addr(&sK[idx]), &Kf[((long)(k0 + r) * NH + h) * DQR + c]);
    }
    // stage V tile transposed (d x 64)
    for (int chunk = tid; chunk < 1024; chunk += 128) {
      int kk = chunk >> 4, dc = (chunk & 15) * 8;
      union { uint4 u; bf16_t hv[8]; } t;
      t.u = *(const uint4*)&V[(long)(k0 + kk) * (NH * DV) + h * DV + dc];
#pragma unroll
      for (int e = 0; e < 8; ++e) sVt[(dc + e) * 64 + kk] = t.hv[e];
    }
    wait_async0();
    __syncthreads();

    // ---- S = Q K^T (16 queries x 64 keys per wave)
    v8f s_acc[4] = {};
    const int rowq = q0 + wid * 16 + laneL;
    const bf16_t* qrow = Q + ((long)rowq * NH + h) * DQR;
#pragma unroll
    for (int s = 0; s < 6; ++s) {
      const int d0 = s * 32;
      Frag fa;
      fa.u[0] = *(const uint4*)(qrow + d0 + hiL * 8);
      fa.u[1] = *(const uint4*)(qrow + d0 + 16 + hiL * 8);
#pragma unroll
      for (int j = 0; j < 4; ++j) {
        int rowk = j * 16 + laneL;
        Frag fb;
        fb.u[0] = *(const uint4*)&sK[rowk * DQR + d0 + hiL * 8];
        fb.u[1] = *(const uint4*)&sK[rowk * DQR + d0 + 16 + hiL * 8];
        s_acc[j] = wmma_bf16(fa.v, fb.v, s_acc[j]);
      }
    }

    // ---- scale + causal mask + online softmax (rows owned by lane half)
    float sv[4][8];
#pragma unroll
    for (int j = 0; j < 4; ++j) {
      int keyg = k0 + j * 16 + laneL;
#pragma unroll
      for (int r = 0; r < 8; ++r) {
        int qg = q0 + wid * 16 + hiL * 8 + r;
        sv[j][r] = (keyg <= qg) ? s_acc[j][r] * scale : -1e30f;
      }
    }
#pragma unroll
    for (int r = 0; r < 8; ++r) {
      float m = fmaxf(fmaxf(sv[0][r], sv[1][r]), fmaxf(sv[2][r], sv[3][r]));
      m = fmaxf(m, __shfl_xor(m, 1, 32));
      m = fmaxf(m, __shfl_xor(m, 2, 32));
      m = fmaxf(m, __shfl_xor(m, 4, 32));
      m = fmaxf(m, __shfl_xor(m, 8, 32));
      float mnew = fmaxf(mstate[r], m);
      float corr = __expf(mstate[r] - mnew);
      mstate[r] = mnew;
      float psum = 0.f;
#pragma unroll
      for (int j = 0; j < 4; ++j) {
        float p = __expf(sv[j][r] - mnew);
        sv[j][r] = p;
        psum += p;
      }
      psum += __shfl_xor(psum, 1, 32);
      psum += __shfl_xor(psum, 2, 32);
      psum += __shfl_xor(psum, 4, 32);
      psum += __shfl_xor(psum, 8, 32);
      lstate[r] = lstate[r] * corr + psum;
#pragma unroll
      for (int dt = 0; dt < 8; ++dt) oacc[dt][r] = oacc[dt][r] * corr;
    }

    // ---- P -> LDS (per-wave region; same-wave DS ordering via dscnt)
#pragma unroll
    for (int j = 0; j < 4; ++j)
#pragma unroll
      for (int r = 0; r < 8; ++r)
        sP[wid][(hiL * 8 + r) * 64 + j * 16 + laneL] = (bf16_t)sv[j][r];
    asm volatile("s_wait_dscnt 0" ::: "memory");

    // ---- O += P V
#pragma unroll
    for (int s = 0; s < 2; ++s) {
      const int kk0 = s * 32;
      Frag fp;
      fp.u[0] = *(const uint4*)&sP[wid][laneL * 64 + kk0 + hiL * 8];
      fp.u[1] = *(const uint4*)&sP[wid][laneL * 64 + kk0 + 16 + hiL * 8];
#pragma unroll
      for (int dt = 0; dt < 8; ++dt) {
        int dr = dt * 16 + laneL;
        Frag fv;
        fv.u[0] = *(const uint4*)&sVt[dr * 64 + kk0 + hiL * 8];
        fv.u[1] = *(const uint4*)&sVt[dr * 64 + kk0 + 16 + hiL * 8];
        oacc[dt] = wmma_bf16(fp.v, fv.v, oacc[dt]);
      }
    }
  }

  // ---- normalize + store bf16
#pragma unroll
  for (int dt = 0; dt < 8; ++dt) {
    int dg = dt * 16 + laneL;
#pragma unroll
    for (int r = 0; r < 8; ++r) {
      int qg = q0 + wid * 16 + hiL * 8 + r;
      O[(long)qg * (NH * DV) + h * DV + dg] = (bf16_t)(oacc[dt][r] / lstate[r]);
    }
  }
}

// ---------------------------------------------------------------------------
// launch
// ---------------------------------------------------------------------------
extern "C" void kernel_launch(void* const* d_in, const int* in_sizes, int n_in,
                              void* d_out, int out_size, void* d_ws, size_t ws_size,
                              hipStream_t stream) {
  const int*   positions = (const int*)d_in[0];
  const float* hidden    = (const float*)d_in[1];
  const float* w_qa      = (const float*)d_in[2];
  const float* gamma_q   = (const float*)d_in[3];
  const float* w_qb      = (const float*)d_in[4];
  const float* w_kva     = (const float*)d_in[5];
  const float* gamma_kv  = (const float*)d_in[6];
  const float* w_kvb     = (const float*)d_in[7];
  const float* w_o       = (const float*)d_in[8];

  char* ws = (char*)d_ws;
  size_t off = 0;
  auto take = [&](size_t bytes) -> void* {
    void* p = ws + off;
    off += (bytes + 255) & ~(size_t)255;
    return p;
  };

  bf16_t* hid_b  = (bf16_t*)take((size_t)T_LEN * HIDDEN * 2);
  bf16_t* wqa_b  = (bf16_t*)take((size_t)HIDDEN * Q_LORA * 2);
  bf16_t* wqb_b  = (bf16_t*)take((size_t)Q_LORA * NH * DQR * 2);
  bf16_t* wkva_b = (bf16_t*)take((size_t)HIDDEN * (KV_LORA + DR) * 2);
  bf16_t* wkvb_b = (bf16_t*)take((size_t)KV_LORA * NH * (DQ + DV) * 2);
  bf16_t* wo_b   = (bf16_t*)take((size_t)NH * DV * HIDDEN * 2);
  float*  qa_f   = (float*)take((size_t)T_LEN * Q_LORA * 4);
  bf16_t* qc_b   = (bf16_t*)take((size_t)T_LEN * Q_LORA * 2);
  float*  big_f  = (float*)take((size_t)T_LEN * NH * (DQ + DV) * 4); // qraw then kvup
  bf16_t* qfull  = (bf16_t*)take((size_t)T_LEN * NH * DQR * 2);
  float*  kvr_f  = (float*)take((size_t)T_LEN * (KV_LORA + DR) * 4);
  bf16_t* kvc_b  = (bf16_t*)take((size_t)T_LEN * KV_LORA * 2);
  bf16_t* kfull  = (bf16_t*)take((size_t)T_LEN * NH * DQR * 2);
  bf16_t* v_b    = (bf16_t*)take((size_t)T_LEN * NH * DV * 2);
  bf16_t* attn_b = (bf16_t*)take((size_t)T_LEN * NH * DV * 2);

  auto cvt = [&](const float* s, bf16_t* d, long n) {
    long t = n / 8;
    cvt_f32_bf16<<<dim3((unsigned)((t + 255) / 256)), dim3(256), 0, stream>>>(s, d, n);
  };
  auto gemm = [&](const bf16_t* A, const bf16_t* B, float* C, int M, int N, int K) {
    gemm_bf16_f32<<<dim3(N / 64, M / 128), dim3(256), 0, stream>>>(A, B, C, M, N, K);
  };

  // bf16 staging of activations + weights
  cvt(hidden, hid_b,  (long)T_LEN * HIDDEN);
  cvt(w_qa,   wqa_b,  (long)HIDDEN * Q_LORA);
  cvt(w_qb,   wqb_b,  (long)Q_LORA * NH * DQR);
  cvt(w_kva,  wkva_b, (long)HIDDEN * (KV_LORA + DR));
  cvt(w_kvb,  wkvb_b, (long)KV_LORA * NH * (DQ + DV));
  cvt(w_o,    wo_b,   (long)NH * DV * HIDDEN);

  // q path
  gemm(hid_b, wqa_b, qa_f, T_LEN, Q_LORA, HIDDEN);
  rmsnorm_bf16_kernel<<<T_LEN, 256, 0, stream>>>(qa_f, gamma_q, qc_b, Q_LORA);
  gemm(qc_b, wqb_b, big_f, T_LEN, NH * DQR, Q_LORA);
  q_finalize_kernel<<<T_LEN, 256, 0, stream>>>(big_f, positions, qfull);

  // kv path
  gemm(hid_b, wkva_b, kvr_f, T_LEN, KV_LORA + DR, HIDDEN);
  kv_finalize_kernel<<<T_LEN, 256, 0, stream>>>(kvr_f, gamma_kv, positions, kvc_b, kfull);
  gemm(kvc_b, wkvb_b, big_f, T_LEN, NH * (DQ + DV), KV_LORA);
  {
    long n = (long)T_LEN * NH * (DQ + DV);
    kvup_finalize_kernel<<<dim3((unsigned)((n + 255) / 256)), dim3(256), 0, stream>>>(
        big_f, kfull, v_b);
  }

  // attention
  mla_attn_kernel<<<dim3(T_LEN / 64, NH), dim3(128), 0, stream>>>(qfull, kfull, v_b, attn_b);

  // output projection -> f32 d_out
  gemm(attn_b, wo_b, (float*)d_out, T_LEN, HIDDEN, NH * DV);
}
